// Net_40690520162873
// MI455X (gfx1250) — compile-verified, matrix operand order checked by reading, not run on hardware
//
#include <hip/hip_runtime.h>
#include <hip/hip_bf16.h>
#include <math.h>

#define NNODES 65536
#define NEDGES 1048576
#define NGRAPH 128
#define NPERG  512
#define NHID   128

typedef __attribute__((ext_vector_type(16))) __bf16 v16bf;
typedef __attribute__((ext_vector_type(8)))  float  v8f;

// ---------------------------------------------------------------------------
// Dense part: C[N,128] = A[N,128](f32) @ W[128,128], W pre-packed bf16
// transposed ([n][k]).  Block = 128 threads = 4 waves.  The 16x128 A tile is
// staged once in LDS (bf16), then each wave computes a 16x32 output slab:
// 2 f32 accumulators, 8 x v_wmma_f32_16x16x32_bf16 (A fragment reused).
// Fragment layouts follow CDNA5 ISA §7.12.2 (wave32).
// ---------------------------------------------------------------------------
__global__ void gemm_x_w(const float* __restrict__ A, const __bf16* __restrict__ Bt,
                         float* __restrict__ C, int nrows)
{
  __shared__ __bf16 As[16][NHID];
  const int t    = threadIdx.x;            // 0..127
  const int lane = t & 31;
  const int wave = t >> 5;                 // 0..3 -> 32-col slab
  const int row0 = blockIdx.x << 4;
  if (row0 >= nrows) return;

  // Cooperative A tile load + f32->bf16 convert: thread -> row t/8, 16 cols.
  {
    int r = t >> 3, cc = (t & 7) << 4;
    const float* ar = A + (size_t)(row0 + r) * NHID + cc;
    #pragma unroll
    for (int i = 0; i < 16; ++i) As[r][cc + i] = (__bf16)ar[i];
  }
  __syncthreads();

  const int mn = lane & 15;                // A-row / B-col within 16-tile
  const int hi = lane >> 4;                // K-half selector (ISA layout)
  const int col0 = wave << 5;

  v8f c0 = {}, c1 = {};
  const __bf16* b0 = Bt + (size_t)(col0 + mn) * NHID;
  const __bf16* b1 = Bt + (size_t)(col0 + 16 + mn) * NHID;

  #pragma unroll
  for (int kk = 0; kk < NHID; kk += 32) {
    v16bf a, v0, v1;
    #pragma unroll
    for (int i = 0; i < 8; ++i) {
      // 16-bit A 16x32: lanes 0-15 K=0..7/16..23, lanes 16-31 K=8..15/24..31
      a[i]      = As[mn][kk + hi * 8 + i];
      a[i + 8]  = As[mn][kk + 16 + hi * 8 + i];
      v0[i]     = b0[kk + hi * 8 + i];
      v0[i + 8] = b0[kk + 16 + hi * 8 + i];
      v1[i]     = b1[kk + hi * 8 + i];
      v1[i + 8] = b1[kk + 16 + hi * 8 + i];
    }
    c0 = __builtin_amdgcn_wmma_f32_16x16x32_bf16(false, a, false, v0, (short)0, c0,
                                                 false, false);
    c1 = __builtin_amdgcn_wmma_f32_16x16x32_bf16(false, a, false, v1, (short)0, c1,
                                                 false, false);
  }
  // C/D layout: lanes 0-15 -> M = r, lanes 16-31 -> M = r + 8
  #pragma unroll
  for (int r = 0; r < 8; ++r) {
    size_t rowoff = (size_t)(row0 + r + 8 * hi) * NHID;
    C[rowoff + col0 + mn]      = c0[r];
    C[rowoff + col0 + 16 + mn] = c1[r];
  }
}

// Pack one 128x128 f32 weight (row-major [k][n]) into bf16 transposed [n][k].
__global__ void pack_w(const float* __restrict__ W, __bf16* __restrict__ Wt)
{
  int i = blockIdx.x * blockDim.x + threadIdx.x;
  if (i >= NHID * NHID) return;
  int k = i >> 7, n = i & 127;
  Wt[n * NHID + k] = (__bf16)W[k * NHID + n];
}

// ---------------------------------------------------------------------------
// Utility fills / copies
// ---------------------------------------------------------------------------
__global__ void fill_f32(float* __restrict__ p, float v, size_t n)
{
  size_t i = (size_t)blockIdx.x * blockDim.x + threadIdx.x;
  if (i < n) p[i] = v;
}
__global__ void copy_f32(float* __restrict__ d, const float* __restrict__ s, size_t n)
{
  size_t i = (size_t)blockIdx.x * blockDim.x + threadIdx.x;
  if (i < n) d[i] = s[i];
}
__global__ void zero_i32(int* __restrict__ p, int n)
{
  int i = blockIdx.x * blockDim.x + threadIdx.x;
  if (i < n) p[i] = 0;
}
__global__ void copy_i32(int* __restrict__ d, const int* __restrict__ s, int n)
{
  int i = blockIdx.x * blockDim.x + threadIdx.x;
  if (i < n) d[i] = s[i];
}
__global__ void set_i32(int* __restrict__ p, int idx, int v)
{
  if (threadIdx.x == 0 && blockIdx.x == 0) p[idx] = v;
}

// ---------------------------------------------------------------------------
// One-time CSR build over static dst[] (counting sort):
//   cnt -> exclusive scan (two-level, 65536 = 256*256) -> scatter edge ids
// ---------------------------------------------------------------------------
__global__ void count_dst(int* __restrict__ cnt, const int* __restrict__ dst)
{
  int e = blockIdx.x * blockDim.x + threadIdx.x;
  if (e < NEDGES) atomicAdd(&cnt[dst[e]], 1);
}

__global__ void scan_block(const int* __restrict__ cnt, int* __restrict__ offs,
                           int* __restrict__ bsum)
{
  __shared__ int s[256];
  int t = threadIdx.x, b = blockIdx.x, i = (b << 8) + t;
  int v = cnt[i];
  s[t] = v;
  __syncthreads();
  for (int d = 1; d < 256; d <<= 1) {
    int add = (t >= d) ? s[t - d] : 0;
    __syncthreads();
    s[t] += add;
    __syncthreads();
  }
  offs[i] = s[t] - v;                 // exclusive
  if (t == 255) bsum[b] = s[t];       // block total
}

__global__ void scan_bsums(const int* __restrict__ bsum, int* __restrict__ bscan)
{
  __shared__ int s[256];
  int t = threadIdx.x;
  int v = bsum[t];
  s[t] = v;
  __syncthreads();
  for (int d = 1; d < 256; d <<= 1) {
    int add = (t >= d) ? s[t - d] : 0;
    __syncthreads();
    s[t] += add;
    __syncthreads();
  }
  bscan[t] = s[t] - v;                // exclusive
}

__global__ void add_bscan(int* __restrict__ offs, const int* __restrict__ bscan)
{
  int i = blockIdx.x * blockDim.x + threadIdx.x;
  if (i < NNODES) offs[i] += bscan[i >> 8];
}

__global__ void csr_fill(const int* __restrict__ dst, int* __restrict__ pos,
                         int* __restrict__ eidx)
{
  int e = blockIdx.x * blockDim.x + threadIdx.x;
  if (e >= NEDGES) return;
  int p = atomicAdd(&pos[dst[e]], 1);
  eidx[p] = e;
}

// ---------------------------------------------------------------------------
// deg/dis via CSR (no atomics): deg = sum in-edge e_alive + n_alive
// ---------------------------------------------------------------------------
__global__ void dis_csr(const int* __restrict__ offs, const int* __restrict__ eidx,
                        const float* __restrict__ e_alive,
                        const float* __restrict__ n_alive, float* __restrict__ dis)
{
  int i = blockIdx.x * blockDim.x + threadIdx.x;
  if (i >= NNODES) return;
  float d = n_alive[i];
  int end = offs[i + 1];
  for (int idx = offs[i]; idx < end; ++idx) d += e_alive[eidx[idx]];
  dis[i] = (d > 0.0f) ? rsqrtf(fmaxf(d, 1e-12f)) : 0.0f;
}

// ---------------------------------------------------------------------------
// Fused GCN aggregate + epilogue via CSR: one wave per dst node, 4 floats/lane
// accumulated in registers (no atomics; h rows are L2-resident gathers).
//   out = relu((sum_e h[src]*dis[s]*dis[d]*e_alive + h*dis^2*na + b) * na) [+res]
// ---------------------------------------------------------------------------
__global__ void gcn_fused(const float* __restrict__ H, const int* __restrict__ offs,
                          const int* __restrict__ eidx, const int* __restrict__ src,
                          const float* __restrict__ e_alive, const float* __restrict__ dis,
                          const float* __restrict__ n_alive, const float* __restrict__ bias,
                          const float* __restrict__ res, float* __restrict__ out)
{
  int node = (blockIdx.x << 3) + (threadIdx.x >> 5);
  if (node >= NNODES) return;
  int lane = threadIdx.x & 31;
  float dnode = dis[node], na = n_alive[node];
  float acc0 = 0.f, acc1 = 0.f, acc2 = 0.f, acc3 = 0.f;
  int beg = offs[node], end = offs[node + 1];
  for (int idx = beg; idx < end; ++idx) {
    int e = eidx[idx];
    float al = e_alive[e];
    if (idx + 1 < end) {                       // hide gather latency
      int e2 = eidx[idx + 1];
      __builtin_prefetch(H + (size_t)src[e2] * NHID, 0, 1);
    }
    if (al == 0.0f) continue;
    int s = src[e];
    float cf = dis[s] * dnode * al;
    const float* hr = H + (size_t)s * NHID + lane * 4;
    acc0 += hr[0] * cf; acc1 += hr[1] * cf;
    acc2 += hr[2] * cf; acc3 += hr[3] * cf;
  }
  size_t base = (size_t)node * NHID + lane * 4;
  float sc = dnode * dnode * na;
  float a[4] = {acc0, acc1, acc2, acc3};
  #pragma unroll
  for (int j = 0; j < 4; ++j) {
    float tv = (a[j] + H[base + j] * sc + bias[lane * 4 + j]) * na;
    tv = fmaxf(tv, 0.0f);
    if (res) tv += res[base + j];
    out[base + j] = tv;
  }
}

// ---------------------------------------------------------------------------
// SAGPool pieces (score GCN is a scalar CSR gather, fully fused)
// ---------------------------------------------------------------------------
__global__ void matvec128(const float* __restrict__ X, const float* __restrict__ Wp,
                          float* __restrict__ outv)
{
  int i = blockIdx.x * blockDim.x + threadIdx.x;
  if (i >= NNODES) return;
  const float* xr = X + (size_t)i * NHID;
  float acc = 0.0f;
  #pragma unroll 8
  for (int c = 0; c < NHID; ++c) acc += xr[c] * Wp[c];
  outv[i] = acc;
}

__global__ void score_csr(const float* __restrict__ hs, const int* __restrict__ offs,
                          const int* __restrict__ eidx, const int* __restrict__ src,
                          const float* __restrict__ e_alive, const float* __restrict__ dis,
                          const float* __restrict__ n_alive, const float* __restrict__ bp,
                          float* __restrict__ score)
{
  int i = blockIdx.x * blockDim.x + threadIdx.x;
  if (i >= NNODES) return;
  float dnode = dis[i], na = n_alive[i], acc = 0.0f;
  int end = offs[i + 1];
  for (int idx = offs[i]; idx < end; ++idx) {
    int e = eidx[idx];
    float al = e_alive[e];
    if (al != 0.0f) {
      int s = src[e];
      acc += hs[s] * dis[s] * dnode * al;
    }
  }
  score[i] = (acc + hs[i] * dnode * dnode * na + bp[0]) * na;
}

// Per-graph exact top-k via bitonic sort of 512 (score,idx) pairs in LDS.
__global__ void topk_kernel(const float* __restrict__ score,
                            const float* __restrict__ n_alive,
                            float* __restrict__ keep, int k)
{
  __shared__ float v[NPERG];
  __shared__ int   id[NPERG];
  int g = blockIdx.x, t = threadIdx.x;
  int node = g * NPERG + t;
  v[t]  = (n_alive[node] > 0.0f) ? score[node] : -__builtin_inff();
  id[t] = t;
  __syncthreads();
  for (int size = 2; size <= NPERG; size <<= 1) {
    for (int stride = size >> 1; stride > 0; stride >>= 1) {
      int p = t ^ stride;
      if (p > t) {
        bool desc = ((t & size) == 0);
        float va = v[t], vb = v[p];
        if (desc ? (va < vb) : (va > vb)) {
          v[t] = vb; v[p] = va;
          int tmp = id[t]; id[t] = id[p]; id[p] = tmp;
        }
      }
      __syncthreads();
    }
  }
  keep[g * NPERG + id[t]] = (t < k) ? 1.0f : 0.0f;
}

__global__ void apply_pool(float* __restrict__ x, const float* __restrict__ score,
                           const float* __restrict__ keep, float* __restrict__ n_alive)
{
  size_t i = (size_t)blockIdx.x * blockDim.x + threadIdx.x;
  if (i >= (size_t)NNODES * NHID) return;
  int node = (int)(i >> 7);
  float m = tanhf(score[node]) * keep[node];
  x[i] *= m;
  if ((i & 127) == 0) n_alive[node] = keep[node];
}

__global__ void edge_mask(float* __restrict__ e_alive, const int* __restrict__ src,
                          const int* __restrict__ dst, const float* __restrict__ keep)
{
  int e = blockIdx.x * blockDim.x + threadIdx.x;
  if (e >= NEDGES) return;
  e_alive[e] *= keep[src[e]] * keep[dst[e]];
}

// Readout: r[g][0:128] += max over alive nodes; r[g][128:256] += sum/k
__global__ void readout_kernel(const float* __restrict__ x, const float* __restrict__ n_alive,
                               float* __restrict__ r, float invk)
{
  int g = blockIdx.x, f = threadIdx.x;   // blockDim = 128
  float mx = -__builtin_inff(), sum = 0.0f;
  for (int n = 0; n < NPERG; ++n) {
    int node = g * NPERG + n;
    float val = x[(size_t)node * NHID + f];
    if (n_alive[node] > 0.0f) mx = fmaxf(mx, val);
    sum += val;   // dead nodes already zeroed by pool gating
  }
  r[g * 256 + f]       += mx;
  r[g * 256 + 128 + f] += sum * invk;
}

// ---------------------------------------------------------------------------
// Final MLP + log-softmax (tiny: 128 rows)
// ---------------------------------------------------------------------------
__global__ void lin_kernel(const float* __restrict__ X, const float* __restrict__ W,
                           const float* __restrict__ b, float* __restrict__ Y,
                           int rows, int fin, int fout, int relu)
{
  int i = blockIdx.x * blockDim.x + threadIdx.x;
  if (i >= rows * fout) return;
  int r = i / fout, o = i % fout;
  float acc = b[o];
  const float* xr = X + (size_t)r * fin;
  for (int c = 0; c < fin; ++c) acc += xr[c] * W[c * fout + o];
  Y[i] = relu ? fmaxf(acc, 0.0f) : acc;
}

__global__ void logsoftmax10(float* __restrict__ y)
{
  int r = blockIdx.x * blockDim.x + threadIdx.x;
  if (r >= NGRAPH) return;
  float* row = y + r * 10;
  float mx = row[0];
  for (int j = 1; j < 10; ++j) mx = fmaxf(mx, row[j]);
  float s = 0.0f;
  for (int j = 0; j < 10; ++j) s += expf(row[j] - mx);
  float lse = mx + logf(s);
  for (int j = 0; j < 10; ++j) row[j] -= lse;
}

// ---------------------------------------------------------------------------
// Host orchestration
// ---------------------------------------------------------------------------
static inline int cdiv(size_t a, int b) { return (int)((a + (size_t)b - 1) / (size_t)b); }

extern "C" void kernel_launch(void* const* d_in, const int* in_sizes, int n_in,
                              void* d_out, int out_size, void* d_ws, size_t ws_size,
                              hipStream_t stream)
{
  (void)in_sizes; (void)n_in; (void)out_size; (void)ws_size;
  const float* x_in = (const float*)d_in[0];
  const float* convW[14]; const float* convB[14];
  for (int j = 0; j < 14; ++j) { convW[j] = (const float*)d_in[1 + 2 * j];
                                 convB[j] = (const float*)d_in[2 + 2 * j]; }
  const float* poolW[3]; const float* poolB[3];
  for (int p = 0; p < 3; ++p) { poolW[p] = (const float*)d_in[29 + 2 * p];
                                poolB[p] = (const float*)d_in[30 + 2 * p]; }
  const float* lin1W = (const float*)d_in[35]; const float* lin1B = (const float*)d_in[36];
  const float* lin2W = (const float*)d_in[37]; const float* lin2B = (const float*)d_in[38];
  const float* lin3W = (const float*)d_in[39]; const float* lin3B = (const float*)d_in[40];
  const int* src = (const int*)d_in[41];
  const int* dst = (const int*)d_in[42];
  float* out = (float*)d_out;

  // Workspace carve-out
  char* w = (char*)d_ws;
  auto alloc = [&](size_t bytes) { void* p = (void*)w; w += (bytes + 255) & ~(size_t)255; return p; };
  const size_t NB = (size_t)NNODES * NHID;
  float* XA = (float*)alloc(NB * 4);
  float* XB = (float*)alloc(NB * 4);
  float* H  = (float*)alloc(NB * 4);
  float* deg_dis = (float*)alloc((size_t)NNODES * 4);   // dis
  float* n_alive = (float*)alloc((size_t)NNODES * 4);
  float* keep    = (float*)alloc((size_t)NNODES * 4);
  float* score   = (float*)alloc((size_t)NNODES * 4);
  float* hs      = (float*)alloc((size_t)NNODES * 4);
  float* e_alive = (float*)alloc((size_t)NEDGES * 4);
  int* cnt   = (int*)alloc((size_t)NNODES * 4);
  int* offs  = (int*)alloc((size_t)(NNODES + 256) * 4);
  int* bsum  = (int*)alloc(256 * 4);
  int* bscan = (int*)alloc(256 * 4);
  int* pos   = (int*)alloc((size_t)NNODES * 4);
  int* eidx  = (int*)alloc((size_t)NEDGES * 4);
  float* R  = (float*)alloc((size_t)NGRAPH * 256 * 4);
  float* M1 = (float*)alloc((size_t)NGRAPH * 128 * 4);
  float* M2 = (float*)alloc((size_t)NGRAPH * 64 * 4);
  __bf16* WT = (__bf16*)alloc((size_t)14 * NHID * NHID * 2);
  float* dis = deg_dis;

  // ---- One-time CSR build over static dst[] -------------------------------
  zero_i32<<<cdiv(NNODES, 256), 256, 0, stream>>>(cnt, NNODES);
  count_dst<<<cdiv(NEDGES, 256), 256, 0, stream>>>(cnt, dst);
  scan_block<<<256, 256, 0, stream>>>(cnt, offs, bsum);
  scan_bsums<<<1, 256, 0, stream>>>(bsum, bscan);
  add_bscan<<<cdiv(NNODES, 256), 256, 0, stream>>>(offs, bscan);
  set_i32<<<1, 1, 0, stream>>>(offs, NNODES, NEDGES);
  copy_i32<<<cdiv(NNODES, 256), 256, 0, stream>>>(pos, offs, NNODES);
  csr_fill<<<cdiv(NEDGES, 256), 256, 0, stream>>>(dst, pos, eidx);

  // ---- Pack conv weights to bf16 transposed -------------------------------
  for (int j = 0; j < 14; ++j)
    pack_w<<<cdiv(NHID * NHID, 256), 256, 0, stream>>>(convW[j], WT + (size_t)j * NHID * NHID);

  // ---- Init ---------------------------------------------------------------
  copy_f32<<<cdiv(NB, 256), 256, 0, stream>>>(XA, x_in, NB);
  fill_f32<<<cdiv(NNODES, 256), 256, 0, stream>>>(n_alive, 1.0f, NNODES);
  fill_f32<<<cdiv(NEDGES, 256), 256, 0, stream>>>(e_alive, 1.0f, NEDGES);
  fill_f32<<<cdiv((size_t)NGRAPH * 256, 256), 256, 0, stream>>>(R, 0.0f, (size_t)NGRAPH * 256);

  auto compute_dis = [&]() {
    dis_csr<<<cdiv(NNODES, 256), 256, 0, stream>>>(offs, eidx, e_alive, n_alive, dis);
  };

  auto run_conv = [&](int j, const float* xin, const float* res, float* xout) {
    gemm_x_w<<<NNODES / 16, 128, 0, stream>>>(xin, WT + (size_t)j * NHID * NHID, H, NNODES);
    gcn_fused<<<NNODES / 8, 256, 0, stream>>>(H, offs, eidx, src, e_alive, dis,
                                              n_alive, convB[j], res, xout);
  };

  auto run_pool = [&](int p, float* x, int k) {
    matvec128<<<cdiv(NNODES, 256), 256, 0, stream>>>(x, poolW[p], hs);
    score_csr<<<cdiv(NNODES, 256), 256, 0, stream>>>(hs, offs, eidx, src, e_alive,
                                                     dis, n_alive, poolB[p], score);
    topk_kernel<<<NGRAPH, NPERG, 0, stream>>>(score, n_alive, keep, k);
    apply_pool<<<cdiv(NB, 256), 256, 0, stream>>>(x, score, keep, n_alive);
    readout_kernel<<<NGRAPH, NHID, 0, stream>>>(x, n_alive, R, 1.0f / (float)k);
    edge_mask<<<cdiv(NEDGES, 256), 256, 0, stream>>>(e_alive, src, dst, keep);
  };

  float* cur = XA; float* nxt = XB;
  auto swp = [&]() { float* t = cur; cur = nxt; nxt = t; };

  // ---- Stage 1: conv1..conv4 (residual on conv2,3), pool K=256
  compute_dis();
  run_conv(0, cur, nullptr, nxt); swp();
  run_conv(1, cur, cur, nxt);     swp();
  run_conv(2, cur, cur, nxt);     swp();
  run_conv(3, cur, nullptr, nxt); swp();
  run_pool(0, cur, 256);

  // ---- Stage 2: conv5..conv9 (residual on conv5..8), pool K=128
  compute_dis();
  run_conv(4, cur, cur, nxt); swp();
  run_conv(5, cur, cur, nxt); swp();
  run_conv(6, cur, cur, nxt); swp();
  run_conv(7, cur, cur, nxt); swp();
  run_conv(8, cur, nullptr, nxt); swp();
  run_pool(1, cur, 128);

  // ---- Stage 3: conv10..conv14 (residual on conv10..13), pool K=64
  compute_dis();
  run_conv(9,  cur, cur, nxt); swp();
  run_conv(10, cur, cur, nxt); swp();
  run_conv(11, cur, cur, nxt); swp();
  run_conv(12, cur, cur, nxt); swp();
  run_conv(13, cur, nullptr, nxt); swp();
  run_pool(2, cur, 64);

  // ---- MLP head: [128,256] -> 128 -> 64 -> 10 -> log_softmax
  lin_kernel<<<cdiv((size_t)NGRAPH * 128, 256), 256, 0, stream>>>(R, lin1W, lin1B, M1, NGRAPH, 256, 128, 1);
  lin_kernel<<<cdiv((size_t)NGRAPH * 64, 256), 256, 0, stream>>>(M1, lin2W, lin2B, M2, NGRAPH, 128, 64, 1);
  lin_kernel<<<cdiv((size_t)NGRAPH * 10, 256), 256, 0, stream>>>(M2, lin3W, lin3B, out, NGRAPH, 64, 10, 0);
  logsoftmax10<<<1, 128, 0, stream>>>(out);
}
